// EWGCN_86543591015294
// MI455X (gfx1250) — matrix operand order repaired and checked by reading, not compiled, not run on hardware
//
#include <hip/hip_runtime.h>

#define HID 64
#define NCLS 20
#define NGRAPH 64
#define MAXKC 10   // max K chunks of 32 (K=300 -> 10)

typedef __attribute__((ext_vector_type(16))) _Float16 v16h;
typedef __attribute__((ext_vector_type(8)))  _Float16 v8h;
typedef __attribute__((ext_vector_type(8)))  float    v8f;
typedef __attribute__((ext_vector_type(4)))  float    v4f;

// ---------------------------------------------------------------- utilities
__global__ void k_fill_f32(float* __restrict__ p, int n, float v) {
  int i = blockIdx.x * blockDim.x + threadIdx.x;
  if (i < n) p[i] = v;
}

__global__ void k_deg_edges(const int* __restrict__ col, float* __restrict__ deg, int E) {
  int e = blockIdx.x * blockDim.x + threadIdx.x;
  if (e < E) atomicAdd(&deg[col[e]], 1.0f);
}

__global__ void k_dinv(const float* __restrict__ deg, float* __restrict__ dinv, int n) {
  int i = blockIdx.x * blockDim.x + threadIdx.x;
  if (i < n) {
    float d = deg[i];
    dinv[i] = (d > 0.f) ? rsqrtf(d) : 0.f;
  }
}

// ---------------------------------------------------------------- B packing
// Repack W[K x 64] (f32) into WMMA B-fragment order, zero-padded to kc chunks:
// Bpk[((kb*4 + ntile)*32 + lane)*16 + j] = f16( W[kb*32 + (lane>>4)*8 + (j>>3)*16 + (j&7)][ntile*16 + (lane&15)] )
__global__ void k_pack_w(const float* __restrict__ W, _Float16* __restrict__ Bpk,
                         int K, int kc) {
  int idx = blockIdx.x * blockDim.x + threadIdx.x;
  if (idx >= kc * 2048) return;
  int j    = idx & 15;
  int lane = (idx >> 4) & 31;
  int n    = (idx >> 9) & 3;
  int kb   = idx >> 11;
  int half = lane >> 4, l16 = lane & 15;
  int k   = kb * 32 + half * 8 + (j >> 3) * 16 + (j & 7);
  int col = n * 16 + l16;
  float f = (k < K) ? W[k * HID + col] : 0.f;
  Bpk[idx] = (_Float16)f;
}

// ---------------------------------------------------------------- WMMA GEMM
// C[M x 64] = A[M x K] * B[K x 64]; A is f32 (converted in-flight) or f16.
// 4 waves/block, each wave owns a 16x64 tile (4 accumulators).
// B comes pre-packed in fragment order and is staged in LDS.
template <typename AT>
__global__ __launch_bounds__(128) void k_gemm_wmma(const AT* __restrict__ A,
                                                   const _Float16* __restrict__ Bpk,
                                                   float* __restrict__ C,
                                                   int M, int K, int kc) {
  __shared__ __align__(16) _Float16 Blds[MAXKC * 2048];

  { // cooperative stage of packed B (kc * 4KB) into LDS
    const uint4* src = (const uint4*)Bpk;
    uint4* dst = (uint4*)Blds;
    const int total = kc * 256;                // uint4 count
    for (int i = threadIdx.x; i < total; i += 128) dst[i] = src[i];
  }
  __syncthreads();

  const int lane = threadIdx.x & 31;
  const int wave = threadIdx.x >> 5;
  const int half = lane >> 4;                  // 0: lanes 0-15, 1: lanes 16-31
  const int l16  = lane & 15;
  const int mbase = blockIdx.x * 64 + wave * 16;
  const int arow  = mbase + l16;
  const int arow_c = (arow < M) ? arow : (M - 1);   // clamp once; stores are guarded
  const long abase = (long)arow_c * K;
  const int kfull = K >> 5;                    // chunks needing no K-guard
  const _Float16* blane = Blds + (lane << 4);

  v8f acc[4] = {};

  // ---------------- full K chunks: branch-free, vector loads only
  int kb = 0;
  for (; kb < kfull; ++kb) {
    const int kbase = kb << 5;
    __builtin_prefetch((const void*)(A + abase + kbase + 32), 0, 1);

    // A fragment: lane holds K spans [8h..8h+7] and [16+8h..16+8h+7]
    union { v16h v; v8h h[2]; } a;
    const AT* ap = A + abase + kbase + 8 * half;
    if constexpr (sizeof(AT) == 2) {
      a.h[0] = *(const v8h*)(ap);
      a.h[1] = *(const v8h*)(ap + 16);
    } else {
      const v4f s0 = *(const v4f*)(ap);
      const v4f s1 = *(const v4f*)(ap + 4);
      const v4f s2 = *(const v4f*)(ap + 16);
      const v4f s3 = *(const v4f*)(ap + 20);
#pragma unroll
      for (int j = 0; j < 4; ++j) {
        a.v[j]      = (_Float16)s0[j];
        a.v[4 + j]  = (_Float16)s1[j];
        a.v[8 + j]  = (_Float16)s2[j];
        a.v[12 + j] = (_Float16)s3[j];
      }
    }

    // load all 4 B fragments first (distinct regs) so ds-loads overlap WMMA
    union { v16h v; v8h h[2]; } b0, b1, b2, b3;
    const _Float16* bp = blane + kb * 2048;
    b0.h[0] = *(const v8h*)(bp);          b0.h[1] = *(const v8h*)(bp + 8);
    b1.h[0] = *(const v8h*)(bp + 512);    b1.h[1] = *(const v8h*)(bp + 520);
    b2.h[0] = *(const v8h*)(bp + 1024);   b2.h[1] = *(const v8h*)(bp + 1032);
    b3.h[0] = *(const v8h*)(bp + 1536);   b3.h[1] = *(const v8h*)(bp + 1544);

    acc[0] = __builtin_amdgcn_wmma_f32_16x16x32_f16(false, a.v, false, b0.v, (short)0, acc[0], false, false);
    acc[1] = __builtin_amdgcn_wmma_f32_16x16x32_f16(false, a.v, false, b1.v, (short)0, acc[1], false, false);
    acc[2] = __builtin_amdgcn_wmma_f32_16x16x32_f16(false, a.v, false, b2.v, (short)0, acc[2], false, false);
    acc[3] = __builtin_amdgcn_wmma_f32_16x16x32_f16(false, a.v, false, b3.v, (short)0, acc[3], false, false);
  }

  // ---------------- single tail chunk (K % 32 != 0): clamp + select, branch-free
  if (kb < kc) {
    const int kbase = kb << 5;
    union { v16h v; v8h h[2]; } a;
#pragma unroll
    for (int j = 0; j < 16; ++j) {
      const int span = j >> 3;
      const int k = kbase + 8 * half + span * 16 + (j & 7);
      const int kcl = (k < K) ? k : (K - 1);
      const float f = (float)A[abase + kcl];
      a.v[j] = (_Float16)((k < K) ? f : 0.f);
    }
    union { v16h v; v8h h[2]; } b0, b1, b2, b3;
    const _Float16* bp = blane + kb * 2048;
    b0.h[0] = *(const v8h*)(bp);          b0.h[1] = *(const v8h*)(bp + 8);
    b1.h[0] = *(const v8h*)(bp + 512);    b1.h[1] = *(const v8h*)(bp + 520);
    b2.h[0] = *(const v8h*)(bp + 1024);   b2.h[1] = *(const v8h*)(bp + 1032);
    b3.h[0] = *(const v8h*)(bp + 1536);   b3.h[1] = *(const v8h*)(bp + 1544);

    acc[0] = __builtin_amdgcn_wmma_f32_16x16x32_f16(false, a.v, false, b0.v, (short)0, acc[0], false, false);
    acc[1] = __builtin_amdgcn_wmma_f32_16x16x32_f16(false, a.v, false, b1.v, (short)0, acc[1], false, false);
    acc[2] = __builtin_amdgcn_wmma_f32_16x16x32_f16(false, a.v, false, b2.v, (short)0, acc[2], false, false);
    acc[3] = __builtin_amdgcn_wmma_f32_16x16x32_f16(false, a.v, false, b3.v, (short)0, acc[3], false, false);
  }

  // ---- store D: VGPR r holds M = half*8 + r, N = l16 within each 16-wide tile
#pragma unroll
  for (int n = 0; n < 4; ++n) {
#pragma unroll
    for (int r = 0; r < 8; ++r) {
      const int m = mbase + half * 8 + r;
      if (m < M) C[(long)m * HID + n * 16 + l16] = acc[n][r];
    }
  }
}

// ---------------------------------------------------------------- aggregation
// self-loop term initializes the accumulator: S[i] = HW[i] * dinv[i]^2
__global__ void k_agg_init(const float* __restrict__ HW, const float* __restrict__ dinv,
                           float* __restrict__ S, int n) {
  int idx = blockIdx.x * blockDim.x + threadIdx.x;
  if (idx < n * HID) {
    float di = dinv[idx >> 6];
    S[idx] = HW[idx] * di * di;
  }
}

// one thread per (edge, 4-float feature chunk): float4 gather + 4 f32 atomics
__global__ void k_agg_edges(const float* __restrict__ HW, const int* __restrict__ row,
                            const int* __restrict__ col, const float* __restrict__ dinv,
                            float* __restrict__ S, int E) {
  int t = blockIdx.x * blockDim.x + threadIdx.x;
  int e = t >> 4;
  if (e >= E) return;
  int c = (t & 15) << 2;
  int r  = row[e];
  int cl = col[e];
  float nrm = dinv[r] * dinv[cl];
  const float4 v = *(const float4*)(HW + (long)r * HID + c);
  float* dst = S + (long)cl * HID + c;
  atomicAdd(dst + 0, v.x * nrm);
  atomicAdd(dst + 1, v.y * nrm);
  atomicAdd(dst + 2, v.z * nrm);
  atomicAdd(dst + 3, v.w * nrm);
}

__global__ void k_relu_bias_to_f16(const float* __restrict__ S, const float* __restrict__ b,
                                   _Float16* __restrict__ Hh, int n) {
  int idx = blockIdx.x * blockDim.x + threadIdx.x;
  if (idx < n * HID) {
    float h = S[idx] + b[idx & 63];
    h = h > 0.f ? h : 0.f;
    Hh[idx] = (_Float16)h;
  }
}

__global__ void k_relu_bias_inplace(float* __restrict__ S, const float* __restrict__ b, int n) {
  int idx = blockIdx.x * blockDim.x + threadIdx.x;
  if (idx < n * HID) {
    float h = S[idx] + b[idx & 63];
    S[idx] = h > 0.f ? h : 0.f;
  }
}

// ---------------------------------------------------------------- pooling + head
__global__ void k_pool_accum(const float* __restrict__ H, const int* __restrict__ batch,
                             const int* __restrict__ nument,
                             float* __restrict__ entSum, float* __restrict__ allSum,
                             float* __restrict__ entCnt, float* __restrict__ nodeCnt, int n) {
  int idx = blockIdx.x * blockDim.x + threadIdx.x;
  if (idx >= n * HID) return;
  int i = idx >> 6, f = idx & 63;
  int g = batch[i];
  float h = H[idx];
  bool ent = (i < nument[g]);
  atomicAdd(&allSum[g * HID + f], h);
  if (ent) atomicAdd(&entSum[g * HID + f], h);
  if (f == 0) {
    atomicAdd(&nodeCnt[g], 1.f);
    if (ent) atomicAdd(&entCnt[g], 1.f);
  }
}

__global__ void k_head(const float* __restrict__ entSum, const float* __restrict__ allSum,
                       const float* __restrict__ entCnt, const float* __restrict__ nodeCnt,
                       const float* __restrict__ Wlin, const float* __restrict__ blin,
                       float* __restrict__ out) {
  int g = threadIdx.x;
  if (g >= NGRAPH) return;
  float ec = entCnt[g];
  float nc = nodeCnt[g];
  bool useEnt = ec > 0.f;
  float pooled[HID];
#pragma unroll 4
  for (int f = 0; f < HID; ++f)
    pooled[f] = useEnt ? entSum[g * HID + f] / (ec + 1e-6f)
                       : allSum[g * HID + f] / nc;
  float logits[NCLS];
  float mx = -3.4e38f;
  for (int c = 0; c < NCLS; ++c) {
    float s = blin[c];
    for (int f = 0; f < HID; ++f) s += pooled[f] * Wlin[f * NCLS + c];
    logits[c] = s;
    mx = fmaxf(mx, s);
  }
  float se = 0.f;
  for (int c = 0; c < NCLS; ++c) se += expf(logits[c] - mx);
  float lse = mx + logf(se);
  for (int c = 0; c < NCLS; ++c) out[g * NCLS + c] = logits[c] - lse;
}

// ---------------------------------------------------------------- launcher
extern "C" void kernel_launch(void* const* d_in, const int* in_sizes, int n_in,
                              void* d_out, int out_size, void* d_ws, size_t ws_size,
                              hipStream_t stream) {
  (void)n_in; (void)out_size; (void)ws_size;
  const float* x      = (const float*)d_in[0];
  const int*   eidx   = (const int*)d_in[1];
  const int*   batch  = (const int*)d_in[2];
  const int*   nument = (const int*)d_in[3];
  const float* W1     = (const float*)d_in[4];
  const float* b1     = (const float*)d_in[5];
  const float* W2     = (const float*)d_in[6];
  const float* b2     = (const float*)d_in[7];
  const float* Wlin   = (const float*)d_in[8];
  const float* blin   = (const float*)d_in[9];
  float* out = (float*)d_out;

  const int IN_DIM = 300;
  const int N = in_sizes[0] / IN_DIM;
  const int E = in_sizes[1] / 2;
  const int* row = eidx;       // edge_index[0] = source
  const int* col = eidx + E;   // edge_index[1] = target
  const int kc1 = (IN_DIM + 31) / 32;   // 10
  const int kc2 = (HID + 31) / 32;      // 2

  // workspace carve-out (256B aligned slabs)
  char* ws = (char*)d_ws;
  size_t off = 0;
  auto alloc = [&](size_t bytes) -> void* {
    off = (off + 255) & ~(size_t)255;
    void* p = ws + off;
    off += bytes;
    return p;
  };
  float*    deg     = (float*)alloc((size_t)N * 4);
  float*    dinv    = (float*)alloc((size_t)N * 4);
  float*    HW      = (float*)alloc((size_t)N * HID * 4);  // GEMM output (both layers)
  float*    S       = (float*)alloc((size_t)N * HID * 4);  // aggregation accumulator
  _Float16* Hh      = (_Float16*)alloc((size_t)N * HID * 2);
  _Float16* B1pk    = (_Float16*)alloc((size_t)kc1 * 2048 * 2);
  _Float16* B2pk    = (_Float16*)alloc((size_t)kc2 * 2048 * 2);
  float*    entSum  = (float*)alloc((size_t)NGRAPH * HID * 4);
  float*    allSum  = (float*)alloc((size_t)NGRAPH * HID * 4);
  float*    entCnt  = (float*)alloc((size_t)NGRAPH * 4);
  float*    nodeCnt = (float*)alloc((size_t)NGRAPH * 4);

  const int T = 256;
  const int gN  = (N + T - 1) / T;
  const int gE  = (E + T - 1) / T;
  const int nh  = N * HID;
  const int gNH = (nh + T - 1) / T;
  const long et = (long)E * 16;
  const int gET = (int)((et + T - 1) / T);
  const int gblk = (N + 63) / 64;

  // degrees / norm
  k_fill_f32<<<gN, T, 0, stream>>>(deg, N, 1.0f);          // self-loop contributes 1
  k_deg_edges<<<gE, T, 0, stream>>>(col, deg, E);
  k_dinv<<<gN, T, 0, stream>>>(deg, dinv, N);

  // pack weights into WMMA fragment order (f16, zero padded)
  k_pack_w<<<(kc1 * 2048 + T - 1) / T, T, 0, stream>>>(W1, B1pk, IN_DIM, kc1);
  k_pack_w<<<(kc2 * 2048 + T - 1) / T, T, 0, stream>>>(W2, B2pk, HID, kc2);

  // layer 1
  k_gemm_wmma<float><<<gblk, 128, 0, stream>>>(x, B1pk, HW, N, IN_DIM, kc1);
  k_agg_init<<<gNH, T, 0, stream>>>(HW, dinv, S, N);
  k_agg_edges<<<gET, T, 0, stream>>>(HW, row, col, dinv, S, E);
  k_relu_bias_to_f16<<<gNH, T, 0, stream>>>(S, b1, Hh, N);

  // layer 2
  k_gemm_wmma<_Float16><<<gblk, 128, 0, stream>>>(Hh, B2pk, HW, N, HID, kc2);
  k_agg_init<<<gNH, T, 0, stream>>>(HW, dinv, S, N);
  k_agg_edges<<<gET, T, 0, stream>>>(HW, row, col, dinv, S, E);
  k_relu_bias_inplace<<<gNH, T, 0, stream>>>(S, b2, N);

  // pooling buffers are contiguous from entSum through nodeCnt
  int zcount = (int)((nodeCnt + NGRAPH) - entSum);
  k_fill_f32<<<(zcount + T - 1) / T, T, 0, stream>>>(entSum, zcount, 0.0f);
  k_pool_accum<<<gNH, T, 0, stream>>>(S, batch, nument, entSum, allSum, entCnt, nodeCnt, N);

  // classifier head + log-softmax
  k_head<<<1, 64, 0, stream>>>(entSum, allSum, entCnt, nodeCnt, Wlin, blin, out);
}
